// LATTE_30683246362741
// MI455X (gfx1250) — compile-verified, matrix operand order checked by reading, not run on hardware
//
#include <hip/hip_runtime.h>

typedef __attribute__((ext_vector_type(2)))  float    v2f;
typedef __attribute__((ext_vector_type(8)))  float    v8f;
typedef __attribute__((ext_vector_type(16))) _Float16 v16h;

#define DD 128   // feature dim
#define MREL 3   // metapaths

// ---------------------------------------------------------------------------
// native fp atomic add (avoid CAS-loop lowering on flat pointers)
// ---------------------------------------------------------------------------
__device__ __forceinline__ void fatomic_add(float* p, float v) {
#if defined(__HIP_DEVICE_COMPILE__)
  (void)unsafeAtomicAdd(p, v);          // lowers to global/flat_atomic_add_f32
#else
  (void)__hip_atomic_fetch_add(p, v, __ATOMIC_RELAXED, __HIP_MEMORY_SCOPE_AGENT);
#endif
}

// ---------------------------------------------------------------------------
// order-preserving float<->uint encode for atomic segment-max via max_u32
// ---------------------------------------------------------------------------
__device__ __forceinline__ unsigned fenc(float f) {
  unsigned u = __float_as_uint(f);
  return (u & 0x80000000u) ? ~u : (u | 0x80000000u);
}
__device__ __forceinline__ float fdec(unsigned u) {
  return (u & 0x80000000u) ? __uint_as_float(u & 0x7FFFFFFFu)
                           : __uint_as_float(~u);
}

// ---------------------------------------------------------------------------
// Projection GEMM:  Y[n,128] = X[n,128] @ W^T + b   (one wave per 16-row strip)
// Uses V_WMMA_F32_16X16X4_F32 (exact f32), 8 col-tiles per strip, K-step 4.
// ---------------------------------------------------------------------------
__global__ void __launch_bounds__(256)
proj_wmma(const float* __restrict__ X, const float* __restrict__ W,
          const float* __restrict__ bias, float* __restrict__ Y, int nrows)
{
  const int lane  = threadIdx.x;                       // 0..31
  const int strip = blockIdx.x * blockDim.y + threadIdx.y;
  const int row0  = strip * 16;
  if (row0 >= nrows) return;                           // wave-uniform exit

  v8f c[8];
#pragma unroll
  for (int t = 0; t < 8; ++t) c[t] = (v8f){0.f,0.f,0.f,0.f,0.f,0.f,0.f,0.f};

  const int mrow = lane & 15;

#if __has_builtin(__builtin_amdgcn_wmma_f32_16x16x4_f32)
  const int kq = (lane >> 4) << 1;                     // 0 or 2 (ISA A/B f32 layout)
#pragma unroll 4
  for (int k0 = 0; k0 < DD; k0 += 4) {
    const float* xp = X + (size_t)(row0 + mrow) * DD + k0 + kq;
    v2f a; a.x = xp[0]; a.y = xp[1];                   // A: 16x4 tile
#pragma unroll
    for (int t = 0; t < 8; ++t) {
      const int col = t * 16 + mrow;                   // B: 4x16 tile of W^T
      const float* wp = W + (size_t)col * DD + k0 + kq;
      v2f b; b.x = wp[0]; b.y = wp[1];
      c[t] = __builtin_amdgcn_wmma_f32_16x16x4_f32(
                 false, a, false, b, (short)0, c[t], false, false);
    }
  }
#else
  // fallback: f16 WMMA (K-step 32) with on-the-fly conversion
  const int k8 = (lane >> 4) << 3;                     // 0 or 8
  for (int k0 = 0; k0 < DD; k0 += 32) {
    v16h a;
#pragma unroll
    for (int p = 0; p < 16; ++p) {
      const int K = ((p >> 3) << 4) + k8 + (p & 7);
      a[p] = (_Float16)X[(size_t)(row0 + mrow) * DD + k0 + K];
    }
#pragma unroll
    for (int t = 0; t < 8; ++t) {
      const int col = t * 16 + mrow;
      v16h b;
#pragma unroll
      for (int p = 0; p < 16; ++p) {
        const int K = ((p >> 3) << 4) + k8 + (p & 7);
        b[p] = (_Float16)W[(size_t)col * DD + k0 + K];
      }
      c[t] = __builtin_amdgcn_wmma_f32_16x16x32_f16(
                 false, a, false, b, (short)0, c[t], false, false);
    }
  }
#endif

  // C/D layout: VGPR i -> M=i (lanes 0-15) / M=i+8 (lanes 16-31), N=lane&15
  const int colIn = lane & 15;
  const int rAdd  = (lane >> 4) << 3;
#pragma unroll
  for (int t = 0; t < 8; ++t) {
    const int colbase = t * 16;
    const float bv = bias[colbase + colIn];
#pragma unroll
    for (int i = 0; i < 8; ++i)
      Y[(size_t)(row0 + rAdd + i) * DD + colbase + colIn] = c[t][i] + bv;
  }
}

// ---------------------------------------------------------------------------
// beta = softmax(x_l @ Wc^T + bc) over 4 relations; one wave per node
// ---------------------------------------------------------------------------
__global__ void __launch_bounds__(256)
beta_kernel(const float* __restrict__ xl, const float* __restrict__ Wc,
            const float* __restrict__ bc, float* __restrict__ beta, int n)
{
  const int lane = threadIdx.x & 31;
  const int node = (int)((blockIdx.x * (unsigned)blockDim.x + threadIdx.x) >> 5);
  if (node >= n) return;
  const float4 x = ((const float4*)(xl + (size_t)node * DD))[lane];
  float logit[4];
#pragma unroll
  for (int rrel = 0; rrel < 4; ++rrel) {
    const float4 w = ((const float4*)(Wc + (size_t)rrel * DD))[lane];
    float d = x.x * w.x + x.y * w.y + x.z * w.z + x.w * w.w;
#pragma unroll
    for (int off = 16; off; off >>= 1) d += __shfl_xor(d, off, 32);
    logit[rrel] = d + bc[rrel];
  }
  if (lane == 0) {
    float mx = fmaxf(fmaxf(logit[0], logit[1]), fmaxf(logit[2], logit[3]));
    float e0 = __expf(logit[0] - mx), e1 = __expf(logit[1] - mx);
    float e2 = __expf(logit[2] - mx), e3 = __expf(logit[3] - mx);
    float inv = 1.f / (e0 + e1 + e2 + e3);
    float4 b4 = make_float4(e0 * inv, e1 * inv, e2 * inv, e3 * inv);
    ((float4*)beta)[node] = b4;
  }
}

// ---------------------------------------------------------------------------
// per-metapath scratch init (amax=-inf encoded, den=0, acc=0); v4 = N*32
// ---------------------------------------------------------------------------
__global__ void __launch_bounds__(256)
fill_kernel(unsigned* __restrict__ amax, float* __restrict__ den,
            float* __restrict__ acc, long long v4)
{
  long long i = (long long)blockIdx.x * blockDim.x + threadIdx.x;
  if (i >= v4) return;
  ((uint4*)amax)[i]  = make_uint4(0x007FFFFFu, 0x007FFFFFu, 0x007FFFFFu, 0x007FFFFFu);
  ((float4*)den)[i]  = make_float4(0.f, 0.f, 0.f, 0.f);
  ((float4*)acc)[i]  = make_float4(0.f, 0.f, 0.f, 0.f);
}

// out = l * beta[:,3]   (self relation), applied before metapath folds
__global__ void __launch_bounds__(256)
init_out_kernel(const float* __restrict__ l, const float* __restrict__ beta,
                float* __restrict__ out, long long v4)
{
  long long i = (long long)blockIdx.x * blockDim.x + threadIdx.x;
  if (i >= v4) return;
  const long long node = i >> 5;
  const float b = beta[node * 4 + 3];
  float4 lv = ((const float4*)l)[i];
  ((float4*)out)[i] = make_float4(lv.x * b, lv.y * b, lv.z * b, lv.w * b);
}

// ---------------------------------------------------------------------------
// edge pass 1: segment max of a = (l[nbr]*al + r[idx]*ar)*s  (one wave/edge)
// ---------------------------------------------------------------------------
__global__ void __launch_bounds__(256)
edge_max_kernel(const long long* __restrict__ ei, const float* __restrict__ l,
                const float* __restrict__ r, const float* __restrict__ al,
                const float* __restrict__ ar, const float* __restrict__ sharpen,
                int m, unsigned* __restrict__ amax, long long E)
{
  const int lane = threadIdx.x & 31;
  const long long e = ((long long)blockIdx.x * blockDim.x + threadIdx.x) >> 5;
  if (e >= E) return;
  const float s = sharpen[m];
  const long long idx = ei[e], nbr = ei[E + e];
  const float4 lv  = ((const float4*)(l + nbr * DD))[lane];
  const float4 rv  = ((const float4*)(r + idx * DD))[lane];
  const float4 alv = ((const float4*)al)[lane];
  const float4 arv = ((const float4*)ar)[lane];
  unsigned* p = amax + idx * DD + lane * 4;
  atomicMax(p + 0, fenc((lv.x * alv.x + rv.x * arv.x) * s));
  atomicMax(p + 1, fenc((lv.y * alv.y + rv.y * arv.y) * s));
  atomicMax(p + 2, fenc((lv.z * alv.z + rv.z * arv.z) * s));
  atomicMax(p + 3, fenc((lv.w * alv.w + rv.w * arv.w) * s));
}

// edge pass 2: den[idx] += exp(a - amax[idx])
__global__ void __launch_bounds__(256)
edge_den_kernel(const long long* __restrict__ ei, const float* __restrict__ l,
                const float* __restrict__ r, const float* __restrict__ al,
                const float* __restrict__ ar, const float* __restrict__ sharpen,
                int m, const unsigned* __restrict__ amax, float* __restrict__ den,
                long long E)
{
  const int lane = threadIdx.x & 31;
  const long long e = ((long long)blockIdx.x * blockDim.x + threadIdx.x) >> 5;
  if (e >= E) return;
  const float s = sharpen[m];
  const long long idx = ei[e], nbr = ei[E + e];
  const float4 lv  = ((const float4*)(l + nbr * DD))[lane];
  const float4 rv  = ((const float4*)(r + idx * DD))[lane];
  const float4 alv = ((const float4*)al)[lane];
  const float4 arv = ((const float4*)ar)[lane];
  const uint4 mu = ((const uint4*)(amax + idx * DD))[lane];
  float* p = den + idx * DD + lane * 4;
  fatomic_add(p + 0, __expf((lv.x * alv.x + rv.x * arv.x) * s - fdec(mu.x)));
  fatomic_add(p + 1, __expf((lv.y * alv.y + rv.y * arv.y) * s - fdec(mu.y)));
  fatomic_add(p + 2, __expf((lv.z * alv.z + rv.z * arv.z) * s - fdec(mu.z)));
  fatomic_add(p + 3, __expf((lv.w * alv.w + rv.w * arv.w) * s - fdec(mu.w)));
}

// edge pass 3: acc[idx] += l[nbr] * exp(a - amax)/ (den + 1e-16)
__global__ void __launch_bounds__(256)
edge_acc_kernel(const long long* __restrict__ ei, const float* __restrict__ l,
                const float* __restrict__ r, const float* __restrict__ al,
                const float* __restrict__ ar, const float* __restrict__ sharpen,
                int m, const unsigned* __restrict__ amax,
                const float* __restrict__ den, float* __restrict__ acc, long long E)
{
  const int lane = threadIdx.x & 31;
  const long long e = ((long long)blockIdx.x * blockDim.x + threadIdx.x) >> 5;
  if (e >= E) return;
  const float s = sharpen[m];
  const long long idx = ei[e], nbr = ei[E + e];
  const float4 lv  = ((const float4*)(l + nbr * DD))[lane];
  const float4 rv  = ((const float4*)(r + idx * DD))[lane];
  const float4 alv = ((const float4*)al)[lane];
  const float4 arv = ((const float4*)ar)[lane];
  const uint4  mu = ((const uint4*)(amax + idx * DD))[lane];
  const float4 dv = ((const float4*)(den + idx * DD))[lane];
  float* p = acc + idx * DD + lane * 4;
  fatomic_add(p + 0, lv.x * (__expf((lv.x*alv.x + rv.x*arv.x)*s - fdec(mu.x)) / (dv.x + 1e-16f)));
  fatomic_add(p + 1, lv.y * (__expf((lv.y*alv.y + rv.y*arv.y)*s - fdec(mu.y)) / (dv.y + 1e-16f)));
  fatomic_add(p + 2, lv.z * (__expf((lv.z*alv.z + rv.z*arv.z)*s - fdec(mu.z)) / (dv.z + 1e-16f)));
  fatomic_add(p + 3, lv.w * (__expf((lv.w*alv.w + rv.w*arv.w)*s - fdec(mu.w)) / (dv.w + 1e-16f)));
}

// out += acc * beta[:,m]; relu on the last fold
__global__ void __launch_bounds__(256)
fold_kernel(const float* __restrict__ acc, const float* __restrict__ beta,
            int m, float* __restrict__ out, int do_relu, long long v4)
{
  long long i = (long long)blockIdx.x * blockDim.x + threadIdx.x;
  if (i >= v4) return;
  const long long node = i >> 5;
  const float b = beta[node * 4 + m];
  float4 a = ((const float4*)acc)[i];
  float4 o = ((float4*)out)[i];
  o.x += a.x * b; o.y += a.y * b; o.z += a.z * b; o.w += a.w * b;
  if (do_relu) {
    o.x = fmaxf(o.x, 0.f); o.y = fmaxf(o.y, 0.f);
    o.z = fmaxf(o.z, 0.f); o.w = fmaxf(o.w, 0.f);
  }
  ((float4*)out)[i] = o;
}

// ---------------------------------------------------------------------------
extern "C" void kernel_launch(void* const* d_in, const int* in_sizes, int n_in,
                              void* d_out, int out_size, void* d_ws, size_t ws_size,
                              hipStream_t stream)
{
  const int N = in_sizes[0] / DD;                       // 100000
  const long long E = (long long)in_sizes[2] / (MREL * 2); // 400000

  const float*     x_l     = (const float*)d_in[0];
  const float*     x_r     = (const float*)d_in[1];
  const long long* ei      = (const long long*)d_in[2];
  const float*     Wl      = (const float*)d_in[3];
  const float*     bl      = (const float*)d_in[4];
  const float*     Wr      = (const float*)d_in[5];
  const float*     br      = (const float*)d_in[6];
  const float*     Wc      = (const float*)d_in[7];
  const float*     bc      = (const float*)d_in[8];
  const float*     attn_l  = (const float*)d_in[9];
  const float*     attn_r  = (const float*)d_in[10];
  const float*     sharpen = (const float*)d_in[11];
  float*           out     = (float*)d_out;

  // workspace carve-up (~258 MB): l, r, beta, amax, den, acc
  float*    l    = (float*)d_ws;
  float*    r    = l + (size_t)N * DD;
  float*    beta = r + (size_t)N * DD;
  unsigned* amax = (unsigned*)(beta + (size_t)N * 4);
  float*    den  = (float*)(amax + (size_t)N * DD);
  float*    acc  = den + (size_t)N * DD;

  // 1) projections via WMMA (one wave per 16-row strip, 8 waves/block)
  const int strips = N / 16;                           // N % 16 == 0
  dim3 pb(32, 8), pg((strips + 7) / 8);
  proj_wmma<<<pg, pb, 0, stream>>>(x_l, Wl, bl, l, N);
  proj_wmma<<<pg, pb, 0, stream>>>(x_r, Wr, br, r, N);

  // 2) relation-mixing weights
  beta_kernel<<<(N + 7) / 8, 256, 0, stream>>>(x_l, Wc, bc, beta, N);

  // 3) self relation initializes out
  const long long v4 = (long long)N * (DD / 4);
  const int fg = (int)((v4 + 255) / 256);
  init_out_kernel<<<fg, 256, 0, stream>>>(l, beta, out, v4);

  // 4) metapaths: segment-softmax + aggregate, folded into out
  const int eg = (int)((E * 32 + 255) / 256);
  for (int m = 0; m < MREL; ++m) {
    const long long* eim = ei + (size_t)m * 2 * E;
    const float* al = attn_l + (size_t)m * DD;
    const float* ar = attn_r + (size_t)m * DD;
    fill_kernel<<<fg, 256, 0, stream>>>(amax, den, acc, v4);
    edge_max_kernel<<<eg, 256, 0, stream>>>(eim, l, r, al, ar, sharpen, m, amax, E);
    edge_den_kernel<<<eg, 256, 0, stream>>>(eim, l, r, al, ar, sharpen, m, amax, den, E);
    edge_acc_kernel<<<eg, 256, 0, stream>>>(eim, l, r, al, ar, sharpen, m, amax, den, acc, E);
    fold_kernel<<<fg, 256, 0, stream>>>(acc, beta, m, out, (m == MREL - 1) ? 1 : 0, v4);
  }
}
